// MoChA_22411139350724
// MI455X (gfx1250) — matrix operand order repaired and checked by reading, not compiled, not run on hardware
//
#include <hip/hip_runtime.h>

// ---------------------------------------------------------------------------
// MoChA forward on MI455X (gfx1250, wave32, WMMA, async global->LDS).
// All GEMMs are TN (A rows of K dot B rows of K) via bf16 WMMA f32-accum.
// ---------------------------------------------------------------------------

typedef __attribute__((ext_vector_type(16))) __bf16 v16bf;
typedef __attribute__((ext_vector_type(8)))  float  v8f;
typedef int v4i_gcc __attribute__((vector_size(16)));  // matches builtin param

#define EPS_F       1e-6f
#define CLAMP_MIN_F 1e-5f
#define SHARP_F     1.0f
#define SCALE_F     0.04419417382415922f   // 1/sqrt(512)
#define KROW_PAD    1504                   // klen rows padded for 16B alignment

#if defined(__has_builtin)
#if __has_builtin(__builtin_amdgcn_global_load_async_to_lds_b128)
#define USE_ASYNC_LDS 1
#endif
#endif

static __device__ __forceinline__ unsigned short f2bf(float f) {
  unsigned u = __builtin_bit_cast(unsigned, f);
  unsigned r = u + 0x7FFFu + ((u >> 16) & 1u);  // round-to-nearest-even
  return (unsigned short)(r >> 16);
}

#ifdef USE_ASYNC_LDS
static __device__ __forceinline__ void async_copy16(const unsigned short* g,
                                                    unsigned short* l) {
  // Generic LDS pointers carry the LDS byte offset in the low 32 bits.
  __builtin_amdgcn_global_load_async_to_lds_b128(
      (__attribute__((address_space(1))) v4i_gcc*)(size_t)g,
      (__attribute__((address_space(3))) v4i_gcc*)(unsigned)(size_t)l, 0, 0);
}
static __device__ __forceinline__ void wait_async0() {
#if __has_builtin(__builtin_amdgcn_s_wait_asynccnt)
  __builtin_amdgcn_s_wait_asynccnt(0);
#else
  asm volatile("s_wait_asynccnt 0" ::: "memory");
#endif
}
#else
static __device__ __forceinline__ void wait_async0() {}
#endif

// ---------------------------------------------------------------------------
// f32 -> bf16 (straight)
// ---------------------------------------------------------------------------
__global__ __launch_bounds__(256) void cvt_bf16_kernel(
    const float* __restrict__ src, unsigned short* __restrict__ dst, long n) {
  long i = (long)blockIdx.x * 256 + threadIdx.x;
  if (i < n) dst[i] = f2bf(src[i]);
}

// ---------------------------------------------------------------------------
// f32 -> bf16 with transpose: src RxC (f32, row-major) -> dst CxR (bf16)
// ---------------------------------------------------------------------------
__global__ __launch_bounds__(256) void cvt_t_bf16_kernel(
    const float* __restrict__ src, unsigned short* __restrict__ dst,
    int R, int C) {
  long i = (long)blockIdx.x * 256 + threadIdx.x;
  long n = (long)R * C;
  if (i < n) {
    int c = (int)(i / R), r = (int)(i - (long)c * R);
    dst[i] = f2bf(src[(size_t)r * C + c]);
  }
}

// ---------------------------------------------------------------------------
// Tiled transpose of v (bf16): v (B,klen,H*dv) -> vT rows (b,h,d) x klen
// (row stride KROW_PAD). 64x64 tiles through LDS.
// ---------------------------------------------------------------------------
__global__ __launch_bounds__(256) void transpose_v_kernel(
    const unsigned short* __restrict__ v, unsigned short* __restrict__ vT,
    int klen, int H, int dv) {
  __shared__ unsigned short tile[64][65];
  const int kb = blockIdx.x * 64;
  const int db = blockIdx.y * 64;
  const int bh = blockIdx.z;
  const int b = bh / H, h = bh - b * H;
  const int t = threadIdx.x;
  const int c = t & 63;
  const int r4 = t >> 6;
  const int HD = H * dv;
  for (int rr = r4; rr < 64; rr += 4) {
    int gk = kb + rr;
    unsigned short val = 0;
    if (gk < klen) val = v[((size_t)b * klen + gk) * HD + h * dv + db + c];
    tile[rr][c] = val;
  }
  __syncthreads();
  for (int rr = r4; rr < 64; rr += 4) {
    int gk = kb + c;
    if (gk < klen)
      vT[((size_t)bh * dv + db + rr) * KROW_PAD + gk] = tile[c][rr];
  }
}

// ---------------------------------------------------------------------------
// Stage a 128-row x 32-K bf16 tile into LDS (rows x 32, row stride 32).
// Fast path: 16B chunks via async global->LDS (or uint4 sync fallback).
// Edge path: guarded scalar with zero-fill.
// ---------------------------------------------------------------------------
static __device__ __forceinline__ void stage_tile(
    const unsigned short* __restrict__ src, long ld, int r0, int k0,
    int Rmax, int Kmax, unsigned short* lds, int t) {
  const bool fast = (r0 + 128 <= Rmax) && (k0 + 32 <= Kmax);
  if (fast) {
#pragma unroll
    for (int i = 0; i < 2; ++i) {
      int c = t + i * 256;               // 512 chunks of 16B
      int row = c >> 2, kc = (c & 3) << 3;
      const unsigned short* g = src + (size_t)(r0 + row) * ld + k0 + kc;
      unsigned short* l = lds + row * 32 + kc;
#ifdef USE_ASYNC_LDS
      async_copy16(g, l);
#else
      *(uint4*)l = *(const uint4*)g;
#endif
    }
  } else {
    for (int idx = t; idx < 128 * 32; idx += 256) {
      int row = idx >> 5, kk = idx & 31;
      int gr = r0 + row, gk = k0 + kk;
      unsigned short val = 0;
      if (gr < Rmax && gk < Kmax) val = src[(size_t)gr * ld + gk];
      lds[idx] = val;
    }
  }
}

// ---------------------------------------------------------------------------
// WMMA bf16 GEMM (TN):  Out = epilogue( A[MxK] . B[NxK]^T )
// MODE 0: +bias, store bf16      MODE 1: *scale + r[h], sigmoid, store f32
// MODE 2: *scale, store f32      MODE 3: store f32
// 256 thr = 8 waves (4M x 2N); block tile 128x128; Ktile 32; double-buffered.
// ---------------------------------------------------------------------------
template <int MODE>
__global__ __launch_bounds__(256) void wmma_gemm_kernel(
    const unsigned short* __restrict__ A, long lda, long aSB, long aSH,
    const unsigned short* __restrict__ Bm, long ldb, long bSB, long bSH,
    void* __restrict__ OutV, long ldo, long oSB, long oSH,
    const float* __restrict__ bias, const float* __restrict__ raddPtr,
    float scale, int M, int N, int Kdim, int H) {
  __shared__ alignas(64) unsigned short ldsA[2][128 * 32];
  __shared__ alignas(64) unsigned short ldsB[2][128 * 32];

  const int t = threadIdx.x;
  const int wave = t >> 5;          // wave32
  const int lane = t & 31;
  const int waveM = wave & 3;       // M offset 0,32,64,96
  const int waveN = wave >> 2;      // N offset 0,64

  const int z = blockIdx.z;
  const int bb = z / H;
  const int hh = z - bb * H;
  A  += (size_t)bb * aSB + (size_t)hh * aSH;
  Bm += (size_t)bb * bSB + (size_t)hh * bSH;

  const int m0 = blockIdx.y * 128;
  const int n0 = blockIdx.x * 128;

  v8f acc[2][4];
#pragma unroll
  for (int i = 0; i < 2; ++i)
#pragma unroll
    for (int j = 0; j < 4; ++j)
#pragma unroll
      for (int r = 0; r < 8; ++r) acc[i][j][r] = 0.0f;

  // prologue: stage first K tile
  stage_tile(A, lda, m0, 0, M, Kdim, ldsA[0], t);
  stage_tile(Bm, ldb, n0, 0, N, Kdim, ldsB[0], t);
  wait_async0();
  __syncthreads();

  int cur = 0;
  for (int k0 = 0; k0 < Kdim; k0 += 32) {
    const int nxt = cur ^ 1;
    if (k0 + 32 < Kdim) {
      stage_tile(A, lda, m0, k0 + 32, M, Kdim, ldsA[nxt], t);
      stage_tile(Bm, ldb, n0, k0 + 32, N, Kdim, ldsB[nxt], t);
    }

    // fragments per documented CDNA5 16-bit layouts
    const unsigned short* cA = ldsA[cur];
    const unsigned short* cB = ldsB[cur];
    v16bf afr[2], bfr[4];
    const int kOffA = (lane < 16) ? 0 : 8;   // A: lanes<16 K{0..7,16..23}
#pragma unroll
    for (int i = 0; i < 2; ++i) {
      const unsigned short* p = cA + (size_t)(waveM * 32 + i * 16 + (lane & 15)) * 32;
      union { v16bf v; uint4 q[2]; } f;
      f.q[0] = *(const uint4*)(p + kOffA);
      f.q[1] = *(const uint4*)(p + kOffA + 16);
      afr[i] = f.v;
    }
    const int kOffB = (lane < 16) ? 0 : 8;   // B: lanes<16 K{0..7,16..23} of col=lane
#pragma unroll
    for (int j = 0; j < 4; ++j) {
      const unsigned short* p = cB + (size_t)(waveN * 64 + j * 16 + (lane & 15)) * 32 + kOffB;
      union { v16bf v; uint4 q[2]; } f;
      f.q[0] = *(const uint4*)(p);
      f.q[1] = *(const uint4*)(p + 16);
      bfr[j] = f.v;
    }
#pragma unroll
    for (int i = 0; i < 2; ++i)
#pragma unroll
      for (int j = 0; j < 4; ++j)
        acc[i][j] = __builtin_amdgcn_wmma_f32_16x16x32_bf16(
            false, afr[i], false, bfr[j], (short)0, acc[i][j], false, false);

    wait_async0();
    __syncthreads();
    cur = nxt;
  }

  // epilogue. C/D layout: VGPR r, lane<16 -> (M=r,N=lane); lane>=16 -> (M=r+8)
  float radd = 0.0f;
  if (MODE == 1) radd = raddPtr ? raddPtr[hh] : 0.0f;
  const int colL = lane & 15;
  const int rowHalf = (lane < 16) ? 0 : 8;

  unsigned short* Ob = (unsigned short*)OutV + (size_t)bb * oSB + (size_t)hh * oSH;
  float* Of = (float*)OutV + (size_t)bb * oSB + (size_t)hh * oSH;

#pragma unroll
  for (int i = 0; i < 2; ++i)
#pragma unroll
    for (int j = 0; j < 4; ++j)
#pragma unroll
      for (int r = 0; r < 8; ++r) {
        int gn = n0 + waveN * 64 + j * 16 + colL;
        int gm = m0 + waveM * 32 + i * 16 + rowHalf + r;
        if (gm < M && gn < N) {
          float v = acc[i][j][r];
          if (MODE == 0) {
            if (bias) v += bias[gn];
            Ob[(size_t)gm * ldo + gn] = f2bf(v);
          } else if (MODE == 1) {
            float e = v * scale + radd;
            Of[(size_t)gm * ldo + gn] = 1.0f / (1.0f + __expf(-e));
          } else if (MODE == 2) {
            Of[(size_t)gm * ldo + gn] = v * scale;
          } else {
            Of[(size_t)gm * ldo + gn] = v;
          }
        }
      }
}

// ---------------------------------------------------------------------------
// Block-wide exclusive prefix of per-thread totals (wave32 shfl + LDS).
// ---------------------------------------------------------------------------
static __device__ __forceinline__ float block_excl_prefix(float total,
                                                          float* ldsW, int t) {
  const int lane = t & 31, w = t >> 5;
  float incl = total;
#pragma unroll
  for (int off = 1; off < 32; off <<= 1) {
    float n = __shfl_up(incl, off, 32);
    if (lane >= off) incl += n;
  }
  __syncthreads();  // protect ldsW reuse from previous call
  if (lane == 31) ldsW[w] = incl;
  __syncthreads();
  float wOff = 0.0f;
  for (int i = 0; i < w; ++i) wOff += ldsW[i];
  return wOff + incl - total;
}

// ---------------------------------------------------------------------------
// Monotonic alignment scan. One block per (b,h); 256 thr x 6 k (1536 >= 1500).
// ---------------------------------------------------------------------------
__global__ __launch_bounds__(256) void alpha_scan_kernel(
    const float* __restrict__ p, float* __restrict__ alpha, int qlen, int klen) {
  __shared__ float ldsW[8];
  const int bh = blockIdx.x;
  const int t = threadIdx.x;
  const float* pBase = p + (size_t)bh * qlen * klen;
  float* aBase = alpha + (size_t)bh * qlen * klen;
  const int KCH = 6;

  float aw[KCH];
#pragma unroll
  for (int j = 0; j < KCH; ++j) aw[j] = (t * KCH + j == 0) ? 1.0f : 0.0f;

  for (int q = 0; q < qlen; ++q) {
    const float* pr = pBase + (size_t)q * klen;
    float pv[KCH], lg[KCH], cp[KCH], dn[KCH], linc[KCH];
    float run = 0.0f;
#pragma unroll
    for (int j = 0; j < KCH; ++j) {
      int k = t * KCH + j;
      float pp = (k < klen) ? pr[k] : 0.0f;
      pv[j] = pp;
      float om = fminf(fmaxf(1.0f - pp, EPS_F), 1.0f);
      lg[j] = __logf(om);
      run += lg[j];
      linc[j] = run;
    }
    float pre = block_excl_prefix(run, ldsW, t);
#pragma unroll
    for (int j = 0; j < KCH; ++j) {
      float s = pre + linc[j];                   // inclusive cumsum(log1mp)
      cp[j] = __expf(s - lg[j]);                 // exclusive cumprod(1-p)
      dn[j] = fminf(fmaxf(cp[j], EPS_F), 1.0f);  // denom
    }
    run = 0.0f;
#pragma unroll
    for (int j = 0; j < KCH; ++j) {
      run += aw[j] / dn[j];
      linc[j] = run;
    }
    pre = block_excl_prefix(run, ldsW, t);
    float* ar = aBase + (size_t)q * klen;
#pragma unroll
    for (int j = 0; j < KCH; ++j) {
      float al = pv[j] * cp[j] * (pre + linc[j]);
      aw[j] = al;
      int k = t * KCH + j;
      if (k < klen) ar[k] = al;
    }
  }
}

// ---------------------------------------------------------------------------
// Chunk softmax + moving sums -> beta (bf16, row stride KROW_PAD).
//   exp_u = clip(exp(u - max), 1e-5); sm[k] = sum exp_u[k-3..k]
//   beta[h][k] = exp_u[k] * sum_{j=k..k+3} alpha[h][j]/sm[j]
// ---------------------------------------------------------------------------
__global__ __launch_bounds__(256) void beta_kernel(
    const float* __restrict__ u, const float* __restrict__ alpha,
    unsigned short* __restrict__ beta, int klen, int qlen, int H) {
  __shared__ float ex[3 + KROW_PAD];
  __shared__ float sm[KROW_PAD];
  __shared__ float tf[KROW_PAD + 3];
  __shared__ float red[8];

  const int bq = blockIdx.x;
  const int b = bq / qlen, q = bq - b * qlen;
  const int t = threadIdx.x;
  const int lane = t & 31, wave = t >> 5;
  const float* ur = u + ((size_t)b * qlen + q) * klen;

  float mx = -3.0e38f;
  for (int k = t; k < klen; k += 256) mx = fmaxf(mx, ur[k]);
#pragma unroll
  for (int off = 16; off > 0; off >>= 1) mx = fmaxf(mx, __shfl_xor(mx, off, 32));
  if (lane == 0) red[wave] = mx;
  __syncthreads();
  float m2 = -3.0e38f;
  for (int i = 0; i < 8; ++i) m2 = fmaxf(m2, red[i]);

  if (t < 3) ex[t] = 0.0f;
  for (int k = t; k < KROW_PAD; k += 256) {
    float e = 0.0f;
    if (k < klen) e = fmaxf(__expf(ur[k] - m2), CLAMP_MIN_F);
    ex[3 + k] = e;
  }
  __syncthreads();
  for (int k = t; k < KROW_PAD; k += 256)
    sm[k] = ex[k] + ex[k + 1] + ex[k + 2] + ex[k + 3];
  __syncthreads();

  for (int h = 0; h < H; ++h) {
    const float* ar = alpha + (((size_t)b * H + h) * qlen + q) * klen;
    for (int k = t; k < KROW_PAD + 3; k += 256) {
      float v = 0.0f;
      if (k < klen) v = ar[k] * SHARP_F / sm[k];
      tf[k] = v;
    }
    __syncthreads();
    unsigned short* br = beta + (((size_t)b * H + h) * qlen + q) * KROW_PAD;
    for (int k = t; k < klen; k += 256) {
      float s = tf[k] + tf[k + 1] + tf[k + 2] + tf[k + 3];
      br[k] = f2bf(ex[3 + k] * s);
    }
    __syncthreads();
  }
}

// ---------------------------------------------------------------------------
// Host launcher
// ---------------------------------------------------------------------------
extern "C" void kernel_launch(void* const* d_in, const int* in_sizes, int n_in,
                              void* d_out, int out_size, void* d_ws,
                              size_t ws_size, hipStream_t stream) {
  (void)in_sizes; (void)n_in; (void)out_size; (void)ws_size;
  const float* key   = (const float*)d_in[0];   // (16,1500,512)
  const float* query = (const float*)d_in[1];   // (16,128,512)
  const float* wk_ma = (const float*)d_in[2];
  const float* bk_ma = (const float*)d_in[3];
  const float* wq_ma = (const float*)d_in[4];
  const float* bq_ma = (const float*)d_in[5];
  const float* rbias = (const float*)d_in[6];   // (4,1,1)
  const float* wk_ca = (const float*)d_in[7];
  const float* bk_ca = (const float*)d_in[8];
  const float* wq_ca = (const float*)d_in[9];
  const float* bq_ca = (const float*)d_in[10];
  const float* wv    = (const float*)d_in[11];

  const int B = 16, KLEN = 1500, QLEN = 128, D = 512, H = 4, DV = 128;
  const long MK = (long)B * KLEN;  // 24000
  const long MQ = (long)B * QLEN;  // 2048

  char* ws = (char*)d_ws;
  auto alloc = [&](size_t bytes) -> char* {
    char* p = ws;
    ws += (bytes + 255) & ~(size_t)255;
    return p;
  };
  unsigned short* key_bf  = (unsigned short*)alloc((size_t)MK * D * 2);
  unsigned short* q_bf    = (unsigned short*)alloc((size_t)MQ * D * 2);
  unsigned short* wkmaT   = (unsigned short*)alloc((size_t)D * D * 2);
  unsigned short* wqmaT   = (unsigned short*)alloc((size_t)D * D * 2);
  unsigned short* wkcaT   = (unsigned short*)alloc((size_t)D * D * 2);
  unsigned short* wqcaT   = (unsigned short*)alloc((size_t)D * D * 2);
  unsigned short* wvT     = (unsigned short*)alloc((size_t)D * D * 2);
  unsigned short* kma_bf  = (unsigned short*)alloc((size_t)MK * D * 2);
  unsigned short* kca_bf  = (unsigned short*)alloc((size_t)MK * D * 2);
  unsigned short* v_bf    = (unsigned short*)alloc((size_t)MK * D * 2);
  unsigned short* vT_bf   = (unsigned short*)alloc((size_t)B * H * DV * KROW_PAD * 2);
  unsigned short* qma_bf  = (unsigned short*)alloc((size_t)MQ * D * 2);
  unsigned short* qca_bf  = (unsigned short*)alloc((size_t)MQ * D * 2);
  float* p_choose = (float*)alloc((size_t)B * H * QLEN * KLEN * 4);
  float* alphaB   = (float*)alloc((size_t)B * H * QLEN * KLEN * 4);
  float* uB       = (float*)alloc((size_t)B * QLEN * KLEN * 4);
  unsigned short* beta_bf = (unsigned short*)alloc((size_t)B * H * QLEN * KROW_PAD * 2);

  auto cvt = [&](const float* s, unsigned short* d2, long n) {
    cvt_bf16_kernel<<<dim3((unsigned)((n + 255) / 256)), 256, 0, stream>>>(s, d2, n);
  };
  auto cvtT = [&](const float* s, unsigned short* d2) {
    cvt_t_bf16_kernel<<<dim3((unsigned)((D * (long)D + 255) / 256)), 256, 0,
                        stream>>>(s, d2, D, D);
  };
  cvt(key, key_bf, MK * D);
  cvt(query, q_bf, MQ * D);
  cvtT(wk_ma, wkmaT);
  cvtT(wq_ma, wqmaT);
  cvtT(wk_ca, wkcaT);
  cvtT(wq_ca, wqcaT);
  cvtT(wv, wvT);

  // ---- projections (MODE 0: +bias, bf16 out; weights pre-transposed NxK) ----
  const unsigned gmK = (unsigned)((MK + 127) / 128);  // 188
  const unsigned gmQ = (unsigned)((MQ + 127) / 128);  // 16
  wmma_gemm_kernel<0><<<dim3(4, gmK, 1), 256, 0, stream>>>(
      key_bf, D, 0, 0, wkmaT, D, 0, 0, kma_bf, D, 0, 0,
      bk_ma, nullptr, 1.0f, (int)MK, D, D, 1);
  wmma_gemm_kernel<0><<<dim3(4, gmK, 1), 256, 0, stream>>>(
      key_bf, D, 0, 0, wkcaT, D, 0, 0, kca_bf, D, 0, 0,
      bk_ca, nullptr, 1.0f, (int)MK, D, D, 1);
  wmma_gemm_kernel<0><<<dim3(4, gmK, 1), 256, 0, stream>>>(
      key_bf, D, 0, 0, wvT, D, 0, 0, v_bf, D, 0, 0,
      nullptr, nullptr, 1.0f, (int)MK, D, D, 1);
  wmma_gemm_kernel<0><<<dim3(4, gmQ, 1), 256, 0, stream>>>(
      q_bf, D, 0, 0, wqmaT, D, 0, 0, qma_bf, D, 0, 0,
      bq_ma, nullptr, 1.0f, (int)MQ, D, D, 1);
  wmma_gemm_kernel<0><<<dim3(4, gmQ, 1), 256, 0, stream>>>(
      q_bf, D, 0, 0, wqcaT, D, 0, 0, qca_bf, D, 0, 0,
      bq_ca, nullptr, 1.0f, (int)MQ, D, D, 1);

  // ---- v transpose: (b,k,h*128+d) -> rows (b,h,d) x klen (stride 1504) ----
  transpose_v_kernel<<<dim3((KLEN + 63) / 64, DV / 64, B * H), 256, 0, stream>>>(
      v_bf, vT_bf, KLEN, H, DV);

  // ---- p_choose = sigmoid(q_ma . k_ma^T * scale + r[h]) ----
  const unsigned gnK = (unsigned)((KLEN + 127) / 128);  // 12
  wmma_gemm_kernel<1><<<dim3(gnK, 1, B * H), 256, 0, stream>>>(
      qma_bf, D, (long)QLEN * D, 128,
      kma_bf, D, (long)KLEN * D, 128,
      p_choose, KLEN, (long)H * QLEN * KLEN, (long)QLEN * KLEN,
      nullptr, rbias, SCALE_F, QLEN, KLEN, 128, H);

  // ---- u = q_ca . k_ca^T * scale  (H_CA = 1, d = 512) ----
  wmma_gemm_kernel<2><<<dim3(gnK, 1, B), 256, 0, stream>>>(
      qca_bf, D, (long)QLEN * D, 0,
      kca_bf, D, (long)KLEN * D, 0,
      uB, KLEN, (long)QLEN * KLEN, 0,
      nullptr, nullptr, SCALE_F, QLEN, KLEN, 512, 1);

  // ---- monotonic alpha scan ----
  alpha_scan_kernel<<<dim3(B * H), 256, 0, stream>>>(p_choose, alphaB, QLEN, KLEN);

  // ---- chunk softmax / moving sums -> beta (bf16, padded rows) ----
  beta_kernel<<<dim3(B * QLEN), 256, 0, stream>>>(uB, alphaB, beta_bf, KLEN, QLEN, H);

  // ---- cv = beta . vT^T -> d_out (B,Q,512) f32 ----
  wmma_gemm_kernel<3><<<dim3(1, 1, B * H), 256, 0, stream>>>(
      beta_bf, KROW_PAD, (long)H * QLEN * KROW_PAD, (long)QLEN * KROW_PAD,
      vT_bf, KROW_PAD, (long)H * DV * KROW_PAD, (long)DV * KROW_PAD,
      d_out, D, (long)QLEN * D, 128,
      nullptr, nullptr, 1.0f, QLEN, 128, KLEN, H);
}
